// StackedBiLSTM_48679159332902
// MI455X (gfx1250) — matrix operand order repaired and checked by reading, not compiled, not run on hardware
//
#include <hip/hip_runtime.h>

typedef __bf16 bf;
typedef __attribute__((ext_vector_type(16))) __bf16 v16bf;
typedef __attribute__((ext_vector_type(8)))  __bf16 v8bf;
typedef __attribute__((ext_vector_type(8)))  float  v8f;

#define DIM   512
#define B_    64
#define T_    512
#define KC    1024          // concat-K for gate GEMMs
#define HSZ   (B_*DIM)      // one hidden-state matrix (64x512)
#define NEG_S 0.001f

// ---------------- WMMA helpers (CDNA5 wave32, 16x16x32 bf16 -> f32) ----------

__device__ __forceinline__ v16bf cat16(v8bf lo, v8bf hi){
  v16bf r;
#pragma unroll
  for(int i=0;i<8;i++){ r[i]=lo[i]; r[i+8]=hi[i]; }
  return r;
}

__device__ __forceinline__ v8f wmma_bf(v16bf a, v16bf b, v8f c){
  // 8 args: (neg_a, A, neg_b, B, c_mod, C, reuse_a, reuse_b)
  return __builtin_amdgcn_wmma_f32_16x16x32_bf16(false, a, false, b, (short)0, c, false, false);
}

// A fragment: 16x32 bf16. lane(0-15)=row M, holds K kb+{0..7,16..23};
// lanes 16-31 same rows, K kb+{8..15,24..31}. rowp = per-lane row pointer.
__device__ __forceinline__ v16bf afrag(const bf* rowp, int kb, int lane){
  const bf* q = rowp + kb + ((lane<16)?0:8);
  return cat16(*(const v8bf*)q, *(const v8bf*)(q+16));
}
// B fragment: 32x16 bf16, column-major storage (column n contiguous in K).
// lane(0-15)=col N holds K kb+{0..7,16..23}; lanes 16-31 hold K kb+{8..15,24..31}.
// colp = per-lane column pointer.
__device__ __forceinline__ v16bf bfrag(const bf* colp, int kb, int lane){
  const bf* q = colp + kb + ((lane<16)?0:8);
  return cat16(*(const v8bf*)q, *(const v8bf*)(q+16));
}

__device__ __forceinline__ float sigm(float x){ return 1.0f/(1.0f+__expf(-x)); }

// ---------------- prep kernels ----------------------------------------------

__global__ void k_cvt_bf(const float* __restrict__ s, bf* __restrict__ d, int n){
  int i = blockIdx.x*blockDim.x + threadIdx.x;
  if(i<n) d[i] = (bf)s[i];
}
__global__ void k_zero_bf(bf* p, int n){
  int i = blockIdx.x*blockDim.x + threadIdx.x;
  if(i<n) p[i] = (bf)0.0f;
}
__global__ void k_zero_f32(float* p, int n){
  int i = blockIdx.x*blockDim.x + threadIdx.x;
  if(i<n) p[i] = 0.0f;
}
// Pack [U;W] per cell into column-major bf16: wg[cell][n*1024 + k]
__global__ void k_pack_gates(const float* __restrict__ U, const float* __restrict__ W,
                             bf* __restrict__ out){
  int i = blockIdx.x*blockDim.x + threadIdx.x;        // 4*2048*1024 total
  if(i >= 4*2048*1024) return;
  int cell = i >> 21;
  int rem  = i & ((1<<21)-1);
  int n    = rem >> 10;
  int k    = rem & 1023;
  float v = (k < 512) ? U[((size_t)cell*512 + k)*2048 + n]
                      : W[((size_t)cell*512 + (k-512))*2048 + n];
  out[i] = (bf)v;
}
__global__ void k_bias_sum(const float* __restrict__ bU, const float* __restrict__ bW,
                           float* __restrict__ out){
  int i = blockIdx.x*blockDim.x + threadIdx.x;
  if(i < 4*2048) out[i] = bU[i] + bW[i];
}

// ---------------- embedding: emb[t*64+b, d] = lrelu(line[b,t,:]@Wemb[d,:]+b) -
// 1x4 N-blocked: one A fragment feeds 4 WMMAs.

__global__ void k_embed(const bf* __restrict__ lineBF, const bf* __restrict__ wemb,
                        const float* __restrict__ bemb, bf* __restrict__ emb){
  int wave = blockIdx.x*(blockDim.x>>5) + (threadIdx.x>>5);   // 16384 waves
  int lane = threadIdx.x & 31;
  int mT = wave >> 3;                  // 2048 M tiles of 16
  int nG = wave & 7;                   // 8 N groups of 64
  int m0 = mT*16, n0 = nG*64;
  int mr = m0 + (lane & 15);           // m = t*64+b
  int t = mr >> 6, b = mr & 63;
  const bf* rowp = lineBF + ((size_t)b*T_ + t)*DIM;   // line (B,T,V) row
  int n_ = lane & 15;
  const bf* c0p = wemb + (size_t)(n0      + n_)*DIM;  // W_emb[d,:] contiguous
  const bf* c1p = wemb + (size_t)(n0 + 16 + n_)*DIM;
  const bf* c2p = wemb + (size_t)(n0 + 32 + n_)*DIM;
  const bf* c3p = wemb + (size_t)(n0 + 48 + n_)*DIM;
  v8f a0 = {}, a1 = {}, a2 = {}, a3 = {};
#pragma unroll 2
  for(int kb=0; kb<DIM; kb+=32){
    v16bf a = afrag(rowp, kb, lane);
    a0 = wmma_bf(a, bfrag(c0p, kb, lane), a0);
    a1 = wmma_bf(a, bfrag(c1p, kb, lane), a1);
    a2 = wmma_bf(a, bfrag(c2p, kb, lane), a2);
    a3 = wmma_bf(a, bfrag(c3p, kb, lane), a3);
  }
  int hi8 = (lane<16)?0:8;
#pragma unroll
  for(int g=0; g<4; g++){
    v8f acc = (g==0)?a0:(g==1)?a1:(g==2)?a2:a3;
    int n = n0 + g*16 + n_;
    float bias = bemb[n];
#pragma unroll
    for(int r=0;r<8;r++){
      float v = acc[r] + bias;
      v = v > 0.0f ? v : NEG_S*v;
      emb[(size_t)(m0 + r + hi8)*DIM + n] = (bf)v;
    }
  }
}

// ---------------- one LSTM cell step (both directions in one launch) ---------
// Only 256 waves run here (latency-critical, ~1 wave/SIMD), so the K-loop is
// software-pipelined: next iteration's 5 fragments are loaded before the
// current iteration's 4 WMMAs, keeping loads in flight behind matrix math.

struct StepArgs {
  const bf*    x[2];     // input rows (64x512 bf16)
  const bf*    hp[2];    // previous hidden (64x512 bf16)
  float*       c[2];     // cell state f32 (in/out)
  const bf*    w[2];     // packed [U;W] col-major (2048 cols x 1024 K)
  const float* bias[2];  // bU+bW (2048)
  bf*          ho[2];    // new hidden out (64x512 bf16)
};

__global__ void k_step(StepArgs A){
  int wave = blockIdx.x*(blockDim.x>>5) + (threadIdx.x>>5);  // 256 waves
  int lane = threadIdx.x & 31;
  int dir  = wave >> 7;        // 128 tile-groups per direction
  int tg   = wave & 127;
  int mT = tg & 3, jT = tg >> 2;
  int m0 = mT*16, j0 = jT*16;

  int mr = m0 + (lane & 15);
  const bf* xrow = A.x[dir]  + (size_t)mr*DIM;
  const bf* hrow = A.hp[dir] + (size_t)mr*DIM;
  int n_ = lane & 15;
  const bf* w = A.w[dir];
  const bf* cf = w + (size_t)(      j0 + n_)*KC;
  const bf* cc = w + (size_t)( 512 + j0 + n_)*KC;
  const bf* ci = w + (size_t)(1024 + j0 + n_)*KC;
  const bf* co = w + (size_t)(1536 + j0 + n_)*KC;

  // warm the weight stream (global_prefetch_b8)
  __builtin_prefetch(cf + 512, 0, 1);
  __builtin_prefetch(cc + 512, 0, 1);
  __builtin_prefetch(ci + 512, 0, 1);
  __builtin_prefetch(co + 512, 0, 1);

  v8f af = {}, ac = {}, ai = {}, ao = {};

  // ---- pipelined K loop: K = [0,512) from x, [512,1024) from h_prev ----
  v16bf a  = afrag(xrow, 0, lane);
  v16bf b0 = bfrag(cf, 0, lane);
  v16bf b1 = bfrag(cc, 0, lane);
  v16bf b2 = bfrag(ci, 0, lane);
  v16bf b3 = bfrag(co, 0, lane);
#pragma unroll
  for(int kb = 32; kb < KC; kb += 32){
    const bf* ap = (kb < 512) ? (xrow + kb) : (hrow + (kb - 512));
    v16bf an = afrag(ap, 0, lane);
    v16bf n0 = bfrag(cf, kb, lane);
    v16bf n1 = bfrag(cc, kb, lane);
    v16bf n2 = bfrag(ci, kb, lane);
    v16bf n3 = bfrag(co, kb, lane);
    af = wmma_bf(a, b0, af);
    ac = wmma_bf(a, b1, ac);
    ai = wmma_bf(a, b2, ai);
    ao = wmma_bf(a, b3, ao);
    a = an; b0 = n0; b1 = n1; b2 = n2; b3 = n3;
  }
  af = wmma_bf(a, b0, af);
  ac = wmma_bf(a, b1, ac);
  ai = wmma_bf(a, b2, ai);
  ao = wmma_bf(a, b3, ao);

  const float* bs = A.bias[dir];
  float bF = bs[j0+n_], bC = bs[512+j0+n_], bI = bs[1024+j0+n_], bO = bs[1536+j0+n_];
  float* cst = A.c[dir];
  bf*    ho  = A.ho[dir];
  int hi8 = (lane<16)?0:8;
#pragma unroll
  for(int r=0;r<8;r++){
    int m = m0 + r + hi8;
    int j = j0 + n_;
    float F  = sigm(af[r] + bF);
    float C1 = tanhf(ac[r] + bC);
    float I  = sigm(ai[r] + bI);
    float O  = sigm(ao[r] + bO);
    size_t idx = (size_t)m*DIM + j;
    float C2 = F*cst[idx] + I*C1;
    cst[idx] = C2;
    ho[idx]  = (bf)(O * tanhf(C2));
  }
}

// ---------------- classifier stage 1: lrelu([LH‖RH] @ W_s1 + b_s1) ----------
// 1x4 N-blocked like k_embed.

__global__ void k_cls1(const bf* __restrict__ HBf, const bf* __restrict__ HBb,
                       const bf* __restrict__ ws1, const float* __restrict__ bs1,
                       bf* __restrict__ hc){
  int wave = blockIdx.x*(blockDim.x>>5) + (threadIdx.x>>5);  // 16352 waves
  int lane = threadIdx.x & 31;
  int mT = wave >> 3;          // 2044 M tiles (M = 511*64 = 32704)
  int nG = wave & 7;           // 8 N groups of 64
  int m0 = mT*16, n0 = nG*64;
  int mr = m0 + (lane & 15);
  int i = mr >> 6, b = mr & 63;           // output time i in [0,511)
  const bf* rowF = HBf + ((size_t)(i+1)*B_ + b)*DIM;     // lh[i+1]
  const bf* rowB = HBb + ((size_t)(511-i)*B_ + b)*DIM;   // rh step 511-i
  int n_ = lane & 15;
  const bf* c0p = ws1 + (size_t)(n0      + n_)*KC;  // W_s1[d,:] contiguous in e
  const bf* c1p = ws1 + (size_t)(n0 + 16 + n_)*KC;
  const bf* c2p = ws1 + (size_t)(n0 + 32 + n_)*KC;
  const bf* c3p = ws1 + (size_t)(n0 + 48 + n_)*KC;
  v8f a0 = {}, a1 = {}, a2 = {}, a3 = {};
#pragma unroll 2
  for(int kb=0; kb<512; kb+=32){
    v16bf a = afrag(rowF, kb, lane);
    a0 = wmma_bf(a, bfrag(c0p, kb, lane), a0);
    a1 = wmma_bf(a, bfrag(c1p, kb, lane), a1);
    a2 = wmma_bf(a, bfrag(c2p, kb, lane), a2);
    a3 = wmma_bf(a, bfrag(c3p, kb, lane), a3);
  }
#pragma unroll 2
  for(int kb=0; kb<512; kb+=32){
    v16bf a = afrag(rowB, kb, lane);
    a0 = wmma_bf(a, bfrag(c0p, 512+kb, lane), a0);
    a1 = wmma_bf(a, bfrag(c1p, 512+kb, lane), a1);
    a2 = wmma_bf(a, bfrag(c2p, 512+kb, lane), a2);
    a3 = wmma_bf(a, bfrag(c3p, 512+kb, lane), a3);
  }
  int hi8 = (lane<16)?0:8;
#pragma unroll
  for(int g=0; g<4; g++){
    v8f acc = (g==0)?a0:(g==1)?a1:(g==2)?a2:a3;
    int n = n0 + g*16 + n_;
    float bias = bs1[n];
#pragma unroll
    for(int r=0;r<8;r++){
      float v = acc[r] + bias;
      v = v > 0.0f ? v : NEG_S*v;
      hc[(size_t)(m0 + r + hi8)*DIM + n] = (bf)v;
    }
  }
}

// ---------------- classifier stage 2: 2 logits + softmax, one wave per row --

__global__ void k_cls2(const bf* __restrict__ hc, const float* __restrict__ ws2,
                       const float* __restrict__ bs2, float* __restrict__ out){
  int wave = blockIdx.x*(blockDim.x>>5) + (threadIdx.x>>5);  // 32704 rows
  int lane = threadIdx.x & 31;
  const bf* row = hc + (size_t)wave*DIM;
  float d0 = 0.0f, d1 = 0.0f;
  int k0 = lane*16;
#pragma unroll
  for(int k=0;k<16;k++){
    float h = (float)row[k0+k];
    d0 += h * ws2[k0+k];
    d1 += h * ws2[512+k0+k];
  }
  for(int off=16; off>0; off>>=1){
    d0 += __shfl_xor(d0, off, 32);
    d1 += __shfl_xor(d1, off, 32);
  }
  if(lane==0){
    float l0 = d0 + bs2[0], l1 = d1 + bs2[1];
    float m = fmaxf(l0, l1);
    float e0 = __expf(l0-m), e1 = __expf(l1-m);
    float s = e0 + e1;
    out[(size_t)wave*2 + 0] = e0/s;
    out[(size_t)wave*2 + 1] = e1/s;
  }
}

// ---------------- host orchestration ----------------------------------------

extern "C" void kernel_launch(void* const* d_in, const int* in_sizes, int n_in,
                              void* d_out, int out_size, void* d_ws, size_t ws_size,
                              hipStream_t stream){
  (void)in_sizes; (void)n_in; (void)out_size; (void)ws_size;
  const float* line = (const float*)d_in[0];
  const float* Wemb = (const float*)d_in[1];
  const float* bemb = (const float*)d_in[2];
  const float* cU   = (const float*)d_in[3];
  const float* cW   = (const float*)d_in[4];
  const float* cbU  = (const float*)d_in[5];
  const float* cbW  = (const float*)d_in[6];
  const float* Ws1  = (const float*)d_in[7];
  const float* bs1  = (const float*)d_in[8];
  const float* Ws2  = (const float*)d_in[9];
  const float* bs2  = (const float*)d_in[10];
  float* out = (float*)d_out;

  auto up = [](size_t x){ return (x + 255) & ~(size_t)255; };
  char* p = (char*)d_ws;
  bf* wembBF = (bf*)p;   p += up((size_t)512*512*sizeof(bf));
  bf* ws1BF  = (bf*)p;   p += up((size_t)512*1024*sizeof(bf));
  bf* wg     = (bf*)p;   p += up((size_t)4*2048*1024*sizeof(bf));
  float* bsum= (float*)p;p += up((size_t)4*2048*sizeof(float));
  bf* lineBF = (bf*)p;   p += up((size_t)B_*T_*DIM*sizeof(bf));
  bf* embB   = (bf*)p;   p += up((size_t)T_*HSZ*sizeof(bf));
  bf* HBf    = (bf*)p;   p += up((size_t)T_*HSZ*sizeof(bf));
  bf* HBb    = (bf*)p;   p += up((size_t)T_*HSZ*sizeof(bf));
  bf* hA     = (bf*)p;   p += (size_t)4*HSZ*sizeof(bf);   // 2 dirs x 2 ping bufs
  bf* hz     = (bf*)p;   p += up((size_t)HSZ*sizeof(bf)); // zero h (contig w/ hA)
  float* cst = (float*)p;p += up((size_t)4*HSZ*sizeof(float)); // cA0 cA1 cB0 cB1
  bf* HC     = (bf*)p;   p += up((size_t)2044*16*DIM*sizeof(bf));

  const size_t WGSZ = (size_t)2048*1024;  // one packed weight matrix (elements)

  // ---- prep (runs every call: deterministic) ----
  k_cvt_bf   <<<(512*512+255)/256, 256, 0, stream>>>(Wemb, wembBF, 512*512);
  k_cvt_bf   <<<(512*1024+255)/256, 256, 0, stream>>>(Ws1, ws1BF, 512*1024);
  k_cvt_bf   <<<(B_*T_*DIM+255)/256, 256, 0, stream>>>(line, lineBF, B_*T_*DIM);
  k_pack_gates<<<(4*2048*1024+255)/256, 256, 0, stream>>>(cU, cW, wg);
  k_bias_sum <<<(4*2048+255)/256, 256, 0, stream>>>(cbU, cbW, bsum);
  k_zero_bf  <<<(5*HSZ+255)/256, 256, 0, stream>>>(hA, 5*HSZ);     // hA bufs + hz
  k_zero_f32 <<<(4*HSZ+255)/256, 256, 0, stream>>>(cst, 4*HSZ);

  // ---- embedding GEMM (16384 waves, 8 per block) ----
  k_embed<<<2048, 256, 0, stream>>>(lineBF, wembBF, bemb, embB);

  // ---- sequential recurrence: 512 steps x (cell A, cell B) ----
  for(int t=0; t<T_; t++){
    int pp = t & 1;
    StepArgs a;
    a.x[0] = embB + (size_t)t*HSZ;          // fwd reads emb[t]
    a.x[1] = embB + (size_t)(T_-1-t)*HSZ;   // bwd reads emb[T-1-t]
    for(int d=0; d<2; d++){
      a.hp[d]   = hA + ((size_t)d*2 + pp)*HSZ;
      a.ho[d]   = hA + ((size_t)d*2 + (pp^1))*HSZ;
      a.c[d]    = cst + (size_t)d*HSZ;
      a.w[d]    = wg + (size_t)(d ? 2 : 0)*WGSZ;   // cells 0 / 2
      a.bias[d] = bsum + (size_t)(d ? 2 : 0)*2048;
    }
    k_step<<<64, 128, 0, stream>>>(a);

    StepArgs b;
    for(int d=0; d<2; d++){
      bf* hist  = d ? HBb : HBf;
      b.x[d]    = a.ho[d];                                 // hA just produced
      b.hp[d]   = t ? (hist + (size_t)(t-1)*HSZ) : hz;     // hB_{t-1}
      b.ho[d]   = hist + (size_t)t*HSZ;                    // hB_t -> history
      b.c[d]    = cst + (size_t)(2 + d)*HSZ;
      b.w[d]    = wg + (size_t)(d ? 3 : 1)*WGSZ;           // cells 1 / 3
      b.bias[d] = bsum + (size_t)(d ? 3 : 1)*2048;
    }
    k_step<<<64, 128, 0, stream>>>(b);
  }

  // ---- classifier ----
  k_cls1<<<2044, 256, 0, stream>>>(HBf, HBb, ws1BF, bs1, HC);  // 16352 waves
  k_cls2<<<4088, 256, 0, stream>>>(HC, Ws2, bs2, out);         // 32704 waves
}
